// DRUCell_49134425866913
// MI455X (gfx1250) — compile-verified, matrix operand order checked by reading, not run on hardware
//
#include <hip/hip_runtime.h>
#include <math.h>

// ---------------- problem dimensions (from reference) ----------------
constexpr int BATCH = 16384;
constexpr int DIN   = 512;
constexpr int DH    = 512;
constexpr int DK    = DIN + DH;   // 1024 (GEMM K)
constexpr int DN1   = 2 * DH;     // 1024 (W_in out features)

// ---------------- tiling ----------------
constexpr int MT  = 128;   // rows per workgroup
constexpr int NT  = 128;   // cols per (half-)block per workgroup
constexpr int KT  = 32;    // K per WMMA step (bf16)
constexpr int LDK = 40;    // padded LDS row stride in bf16 elems (80 B, 16B-aligned)

typedef __attribute__((ext_vector_type(16))) __bf16 bf16x16;
typedef __attribute__((ext_vector_type(8)))  __bf16 bf16x8;
typedef __attribute__((ext_vector_type(8)))  float  v8f;

// Load one lane's 16x32 bf16 WMMA fragment from an LDS tile stored row-major
// with stride LDK.  Per ISA layout: lane holds 8 contiguous K at base
// (lane<16 ? 0 : 8) in v0..3 and base+16 in v4..7.
__device__ inline bf16x16 frag_ld(const __bf16* tile, int row, int lane) {
    const int kbase = (lane >> 4) << 3;          // 0 or 8
    const __bf16* p = tile + row * LDK + kbase;
    bf16x16 f;
    ((bf16x8*)&f)[0] = *(const bf16x8*)(p);      // K = base .. base+7
    ((bf16x8*)&f)[1] = *(const bf16x8*)(p + 16); // K = base+16 .. base+23
    return f;
}

// ---- staging helpers (register-prefetch software pipeline) ----
// A tile: 128 rows x 32 K of concat(x,h); K-tiles never straddle the 512 split.
__device__ inline void a_load(const float* __restrict__ x,
                              const float* __restrict__ h,
                              int m0, int kb, int tid, float4 (&pa)[4]) {
    const float* Asrc = (kb < DIN) ? (x + (size_t)m0 * DIN + kb)
                                   : (h + (size_t)m0 * DH + (kb - DIN));
#pragma unroll
    for (int i = 0; i < 4; ++i) {
        int lin = tid + i * 256;              // 0..1023
        int row = lin >> 3;                   // 0..127
        int c4  = (lin & 7) << 2;             // 0..28
        pa[i] = *(const float4*)(Asrc + (size_t)row * DIN + c4);
    }
}
__device__ inline void a_store(__bf16* lA, int tid, const float4 (&pa)[4]) {
#pragma unroll
    for (int i = 0; i < 4; ++i) {
        int lin = tid + i * 256;
        int row = lin >> 3;
        int c4  = (lin & 7) << 2;
        __bf16* d = &lA[row * LDK + c4];
        d[0] = (__bf16)pa[i].x; d[1] = (__bf16)pa[i].y;
        d[2] = (__bf16)pa[i].z; d[3] = (__bf16)pa[i].w;
    }
}
// B tile: 32 K x 128 N slice of a [K, ldw] weight matrix.
__device__ inline void b_load(const float* __restrict__ W, int ldw,
                              int kb, int n0, int tid, float4 (&pb)[4]) {
    const float* Bsrc = W + (size_t)kb * ldw + n0;
#pragma unroll
    for (int i = 0; i < 4; ++i) {
        int lin = tid + i * 256;              // 0..1023
        int kk  = lin >> 5;                   // 0..31
        int n4  = (lin & 31) << 2;            // 0..124
        pb[i] = *(const float4*)(Bsrc + (size_t)kk * ldw + n4);
    }
}
// Store transposed into LDS: lB[n][k] so B fragments are contiguous along K.
__device__ inline void b_store(__bf16* lB, int tid, const float4 (&pb)[4]) {
#pragma unroll
    for (int i = 0; i < 4; ++i) {
        int lin = tid + i * 256;
        int kk  = lin >> 5;
        int n4  = (lin & 31) << 2;
        lB[(n4 + 0) * LDK + kk] = (__bf16)pb[i].x;
        lB[(n4 + 1) * LDK + kk] = (__bf16)pb[i].y;
        lB[(n4 + 2) * LDK + kk] = (__bf16)pb[i].z;
        lB[(n4 + 3) * LDK + kk] = (__bf16)pb[i].w;
    }
}

// ---- branch-free activations (avoid ocml tanhf's EXEC-branch slow path) ----
__device__ inline float fast_tanh(float v) {
    float e = __expf(2.0f * v);               // +inf / 0 saturate correctly
    return 1.0f - __fdividef(2.0f, e + 1.0f);
}
__device__ inline float fast_sigmoid(float v) {
    return __fdividef(1.0f, 1.0f + __expf(-v));
}

// ====================================================================
// Kernel 1: lin = [x|h] @ W_in + b_in ; gates ; h_new
// Each workgroup: rows [m0,m0+128), cols [n0,n0+128) of BOTH halves
// (learn cols n0.., forget cols DH+n0..) so the gate blend is local.
// ====================================================================
__global__ __launch_bounds__(256) void dru_gate_kernel(
    const float* __restrict__ x, const float* __restrict__ h,
    const float* __restrict__ W_in, const float* __restrict__ b_in,
    float* __restrict__ h_new)
{
    __shared__ alignas(16) __bf16 lA[MT * LDK];          // 10240 B
    __shared__ alignas(16) __bf16 lB[2 * NT * LDK];      // 20480 B

    const int m0   = blockIdx.x * MT;
    const int n0   = blockIdx.y * NT;        // column inside H
    const int tid  = threadIdx.x;
    const int lane = tid & 31;
    const int wave = tid >> 5;               // wave32
    const int wm   = wave >> 2;              // 0..1 : 64-row strip
    const int wn   = wave & 3;               // 0..3 : 32-col strip

    v8f zero = {};
    v8f accL[4][2], accF[4][2];
#pragma unroll
    for (int i = 0; i < 4; ++i)
#pragma unroll
        for (int j = 0; j < 2; ++j) { accL[i][j] = zero; accF[i][j] = zero; }

    // ---- prologue: stage K-tile 0 ----
    float4 pa[4], pbL[4], pbF[4];
    a_load(x, h, m0, 0, tid, pa);
    b_load(W_in, DN1, 0, n0, tid, pbL);
    b_load(W_in, DN1, 0, n0 + DH, tid, pbF);
    a_store(lA, tid, pa);
    b_store(lB, tid, pbL);
    b_store(lB + NT * LDK, tid, pbF);
    __syncthreads();

    for (int kb = 0; kb < DK; kb += KT) {
        const int kn = kb + KT;
        // ---- prefetch next K-tile into registers (no wait yet) ----
        if (kn < DK) {
            a_load(x, h, m0, kn, tid, pa);
            b_load(W_in, DN1, kn, n0, tid, pbL);
            b_load(W_in, DN1, kn, n0 + DH, tid, pbF);
        }
        // ---- compute from LDS ----
        const int rsel = lane & 15;
        bf16x16 bfrL[2], bfrF[2];
#pragma unroll
        for (int ni = 0; ni < 2; ++ni) {
            bfrL[ni] = frag_ld(lB,            wn * 32 + ni * 16 + rsel, lane);
            bfrF[ni] = frag_ld(lB + NT * LDK, wn * 32 + ni * 16 + rsel, lane);
        }
#pragma unroll
        for (int mi = 0; mi < 4; ++mi) {
            bf16x16 afr = frag_ld(lA, wm * 64 + mi * 16 + rsel, lane);
#pragma unroll
            for (int ni = 0; ni < 2; ++ni) {
                accL[mi][ni] = __builtin_amdgcn_wmma_f32_16x16x32_bf16(
                    false, afr, false, bfrL[ni], (short)0, accL[mi][ni], false, false);
                accF[mi][ni] = __builtin_amdgcn_wmma_f32_16x16x32_bf16(
                    false, afr, false, bfrF[ni], (short)0, accF[mi][ni], false, false);
            }
        }
        __syncthreads();                       // everyone done reading LDS
        if (kn < DK) {
            a_store(lA, tid, pa);              // cvt f32->bf16 + ds_store
            b_store(lB, tid, pbL);
            b_store(lB + NT * LDK, tid, pbF);
        }
        __syncthreads();                       // staging visible to all
    }

    // ---- epilogue: gates + blend, write h_new ----
    const int rowHalf = (lane >> 4) << 3;  // C/D layout: lanes 16-31 hold M+8
    const int nc      = lane & 15;
#pragma unroll
    for (int mi = 0; mi < 4; ++mi)
#pragma unroll
        for (int ni = 0; ni < 2; ++ni) {
            int col = n0 + wn * 32 + ni * 16 + nc;
            float bl  = b_in[col];
            float bfb = b_in[col + DH];
#pragma unroll
            for (int r = 0; r < 8; ++r) {
                int row = m0 + wm * 64 + mi * 16 + rowHalf + r;
                float learn = fast_tanh(accL[mi][ni][r] + bl);
                float sig   = fast_sigmoid(accF[mi][ni][r] + bfb);
                float hv    = h[(size_t)row * DH + col];
                h_new[(size_t)row * DH + col] = sig * hv + (1.0f - sig) * learn;
            }
        }
}

// ====================================================================
// Kernel 2: out = tanh([x|h_new] @ W_out + b_out)
// ====================================================================
__global__ __launch_bounds__(256) void dru_out_kernel(
    const float* __restrict__ x, const float* __restrict__ h_new,
    const float* __restrict__ W_out, const float* __restrict__ b_out,
    float* __restrict__ out)
{
    __shared__ alignas(16) __bf16 lA[MT * LDK];
    __shared__ alignas(16) __bf16 lB[NT * LDK];

    const int m0   = blockIdx.x * MT;
    const int n0   = blockIdx.y * NT;
    const int tid  = threadIdx.x;
    const int lane = tid & 31;
    const int wave = tid >> 5;
    const int wm   = wave >> 2;
    const int wn   = wave & 3;

    v8f zero = {};
    v8f acc[4][2];
#pragma unroll
    for (int i = 0; i < 4; ++i)
#pragma unroll
        for (int j = 0; j < 2; ++j) acc[i][j] = zero;

    float4 pa[4], pb[4];
    a_load(x, h_new, m0, 0, tid, pa);
    b_load(W_out, DH, 0, n0, tid, pb);
    a_store(lA, tid, pa);
    b_store(lB, tid, pb);
    __syncthreads();

    for (int kb = 0; kb < DK; kb += KT) {
        const int kn = kb + KT;
        if (kn < DK) {
            a_load(x, h_new, m0, kn, tid, pa);
            b_load(W_out, DH, kn, n0, tid, pb);
        }
        const int rsel = lane & 15;
        bf16x16 bfr[2];
#pragma unroll
        for (int ni = 0; ni < 2; ++ni)
            bfr[ni] = frag_ld(lB, wn * 32 + ni * 16 + rsel, lane);
#pragma unroll
        for (int mi = 0; mi < 4; ++mi) {
            bf16x16 afr = frag_ld(lA, wm * 64 + mi * 16 + rsel, lane);
#pragma unroll
            for (int ni = 0; ni < 2; ++ni)
                acc[mi][ni] = __builtin_amdgcn_wmma_f32_16x16x32_bf16(
                    false, afr, false, bfr[ni], (short)0, acc[mi][ni], false, false);
        }
        __syncthreads();
        if (kn < DK) {
            a_store(lA, tid, pa);
            b_store(lB, tid, pb);
        }
        __syncthreads();
    }

    const int rowHalf = (lane >> 4) << 3;
    const int nc      = lane & 15;
#pragma unroll
    for (int mi = 0; mi < 4; ++mi)
#pragma unroll
        for (int ni = 0; ni < 2; ++ni) {
            int col = n0 + wn * 32 + ni * 16 + nc;
            float bb = b_out[col];
#pragma unroll
            for (int r = 0; r < 8; ++r) {
                int row = m0 + wm * 64 + mi * 16 + rowHalf + r;
                out[(size_t)row * DH + col] = fast_tanh(acc[mi][ni][r] + bb);
            }
        }
}

// ====================================================================
extern "C" void kernel_launch(void* const* d_in, const int* in_sizes, int n_in,
                              void* d_out, int out_size, void* d_ws, size_t ws_size,
                              hipStream_t stream) {
    (void)in_sizes; (void)n_in; (void)out_size; (void)d_ws; (void)ws_size;
    const float* x     = (const float*)d_in[0];
    const float* h     = (const float*)d_in[1];
    const float* W_in  = (const float*)d_in[2];
    const float* b_in  = (const float*)d_in[3];
    const float* W_out = (const float*)d_in[4];
    const float* b_out = (const float*)d_in[5];

    float* out   = (float*)d_out;                       // [B, H]
    float* h_new = out + (size_t)BATCH * DH;            // [B, H] (second tuple elem)

    dim3 grid(BATCH / MT, DH / NT);                     // 128 x 4
    dim3 block(256);
    dru_gate_kernel<<<grid, block, 0, stream>>>(x, h, W_in, b_in, h_new);
    dru_out_kernel <<<grid, block, 0, stream>>>(x, h_new, W_out, b_out, out);
}
